// MSA_37701222924584
// MI455X (gfx1250) — matrix operand order repaired and checked by reading, not compiled, not run on hardware
//
#include <hip/hip_runtime.h>

// ---------------------------------------------------------------------------
// Types for CDNA5 WMMA (wave32): bf16 A/B fragments (16 bf16 = 8 VGPRs/lane),
// f32 C/D fragments (8 f32 = 8 VGPRs/lane).
// ---------------------------------------------------------------------------
typedef __bf16 bf16;
typedef __attribute__((ext_vector_type(4)))  float  v4f;
typedef __attribute__((ext_vector_type(8)))  float  v8f;
typedef __attribute__((ext_vector_type(4)))  __bf16 v4bf;
typedef __attribute__((ext_vector_type(8)))  __bf16 v8bf;
typedef __attribute__((ext_vector_type(16))) __bf16 v16bf;
typedef __attribute__((ext_vector_type(4)))  int    v4i;

union Frag { v16bf v; v8bf h[2]; };

__device__ __forceinline__ v8f wmma_bf16(const v16bf& a, const v16bf& b, const v8f& c) {
  // D = A(16x32 bf16) * B(32x16 bf16) + C(16x16 f32)
  return __builtin_amdgcn_wmma_f32_16x16x32_bf16(false, a, false, b, (short)0, c, false, false);
}

__device__ __forceinline__ v8f vzero8() {
  v8f z;
#pragma unroll
  for (int i = 0; i < 8; ++i) z[i] = 0.f;
  return z;
}

// ---------------------------------------------------------------------------
// Async global->LDS staging (CDNA5 GLOBAL_LOAD_ASYNC_TO_LDS_B128, ASYNCcnt).
// Signature (from clang diagnostic): param0 = int4 in AS1 (global), so we
// cast to the exact pointer types. Falls back to load+ds_store otherwise.
// ---------------------------------------------------------------------------
#if defined(__HIP_DEVICE_COMPILE__) && __has_builtin(__builtin_amdgcn_global_load_async_to_lds_b128)
#define HAVE_ASYNC_LDS 1
#else
#define HAVE_ASYNC_LDS 0
#endif

__device__ __forceinline__ void stage16B(const bf16* __restrict__ g, bf16* l) {
#if HAVE_ASYNC_LDS
  typedef __attribute__((address_space(1))) v4i* gvp_t;
  typedef __attribute__((address_space(3))) v4i* lvp_t;
  bf16* gnc = const_cast<bf16*>(g);
  __builtin_amdgcn_global_load_async_to_lds_b128((gvp_t)gnc, (lvp_t)l, 0, 0);
#else
  *(v8bf*)l = *(const v8bf*)g;
#endif
}

__device__ __forceinline__ void stage_wait() {
#if HAVE_ASYNC_LDS
#if __has_builtin(__builtin_amdgcn_s_wait_asynccnt)
  __builtin_amdgcn_s_wait_asynccnt(0);
#else
  asm volatile("s_wait_asynccnt 0x0" ::: "memory");
#endif
#endif
}

// Problem constants
#define NBATCH   16
#define NTOK     1024
#define DIMC     768
#define NHEADS   12
#define DHEAD    64
#define NBH      (NBATCH * NHEADS)      // 192
#define HEADSZ   (NTOK * DHEAD)         // 65536
#define TOTROWS  (NBATCH * NTOK)        // 16384
#define QKVCOLS  (3 * DIMC)             // 2304
#define ATTN_SCALE 0.125f               // 64^-0.5

// ---------------------------------------------------------------------------
// Kernel 0: transpose fp32 [R,C] -> bf16 [C,R]  (weights prep, tiny)
// ---------------------------------------------------------------------------
__global__ void k_transpose_bf16(const float* __restrict__ src,
                                 bf16* __restrict__ dst, int R, int C) {
  int idx = blockIdx.x * 256 + threadIdx.x;
  if (idx >= R * C) return;
  int r = idx / C;
  int c = idx - r * C;
  dst[(size_t)c * R + r] = (bf16)src[idx];
}

// ---------------------------------------------------------------------------
// Kernel 1: QKV GEMM.  C[16384,2304] = X[16384,768] @ Wqkv
//   A: fp32 global, converted to bf16 while staging to LDS (VGPR path).
//   B: pre-transposed bf16 WqkvT [2304,768] — async-staged to LDS.
//   Tiles: 128x128x32, 256 threads = 8 waves, wave tile 32(m) x 64(n).
//   Epilogue scatters into Q[b,h,n,d]*scale, K[b,h,n,d], Vt[b,h,d,n] (bf16).
// ---------------------------------------------------------------------------
#define LDT 40  // padded LDS stride (bf16 units): 80B rows, conflict-free

__global__ __launch_bounds__(256)
void k_qkv_gemm(const float* __restrict__ X, const bf16* __restrict__ Wt,
                bf16* __restrict__ Qb, bf16* __restrict__ Kb,
                bf16* __restrict__ Vt) {
  __shared__ alignas(16) bf16 As[128 * LDT];
  __shared__ alignas(16) bf16 Bs[128 * LDT];

  const int t    = threadIdx.x;
  const int lane = t & 31, w = t >> 5;
  const int wm   = w & 3, wn = w >> 2;        // 4x2 wave grid
  const int r16  = lane & 15, hf = lane >> 4;
  const int blockRow = blockIdx.x * 128;
  const int blockCol = blockIdx.y * 128;

  v8f acc[2][4];
#pragma unroll
  for (int mi = 0; mi < 2; ++mi)
#pragma unroll
    for (int ni = 0; ni < 4; ++ni) acc[mi][ni] = vzero8();

  for (int kk = 0; kk < DIMC / 32; ++kk) {
    const int k0 = kk * 32;
    // Stage B (bf16 [outcol][k]) via async path: 512 16B chunks / 256 = 2 each.
#pragma unroll
    for (int i = 0; i < 2; ++i) {
      int id = t + 256 * i;
      int n  = id >> 2;
      int c8 = (id & 3) * 8;
      stage16B(Wt + (size_t)(blockCol + n) * DIMC + k0 + c8, &Bs[n * LDT + c8]);
    }
    // Stage A: 128x32 fp32 -> bf16 (needs conversion; register path).
#pragma unroll
    for (int i = 0; i < 4; ++i) {
      int id = t + 256 * i;
      int r  = id >> 3;
      int c4 = (id & 7) * 4;
      v4f f = *(const v4f*)(X + (size_t)(blockRow + r) * DIMC + k0 + c4);
      v4bf o;
      o[0] = (bf16)f[0]; o[1] = (bf16)f[1]; o[2] = (bf16)f[2]; o[3] = (bf16)f[3];
      *(v4bf*)&As[r * LDT + c4] = o;
    }
    stage_wait();
    __syncthreads();

    Frag a[2], b[4];
#pragma unroll
    for (int mi = 0; mi < 2; ++mi) {
      int row = wm * 32 + mi * 16 + r16;
      a[mi].h[0] = *(const v8bf*)&As[row * LDT + 8 * hf];
      a[mi].h[1] = *(const v8bf*)&As[row * LDT + 16 + 8 * hf];
    }
#pragma unroll
    for (int ni = 0; ni < 4; ++ni) {
      int col = wn * 64 + ni * 16 + r16;
      b[ni].h[0] = *(const v8bf*)&Bs[col * LDT + 8 * hf];
      b[ni].h[1] = *(const v8bf*)&Bs[col * LDT + 16 + 8 * hf];
    }
#pragma unroll
    for (int mi = 0; mi < 2; ++mi)
#pragma unroll
      for (int ni = 0; ni < 4; ++ni)
        acc[mi][ni] = wmma_bf16(a[mi].v, b[ni].v, acc[mi][ni]);
    __syncthreads();
  }

  // Epilogue: scatter into per-head Q/K/Vt layouts.
#pragma unroll
  for (int mi = 0; mi < 2; ++mi) {
#pragma unroll
    for (int ni = 0; ni < 4; ++ni) {
      int gc    = blockCol + wn * 64 + ni * 16 + r16;
      int which = gc / DIMC;            // 0=q 1=k 2=v  (constant per fragment)
      int inner = gc - which * DIMC;
      int hh    = inner >> 6;
      int d     = inner & 63;
#pragma unroll
      for (int r = 0; r < 8; ++r) {
        int gr  = blockRow + wm * 32 + mi * 16 + r + 8 * hf;
        int bb  = gr >> 10;
        int tok = gr & 1023;
        size_t bh = (size_t)(bb * NHEADS + hh);
        float v = acc[mi][ni][r];
        if (which == 0)
          Qb[bh * HEADSZ + (size_t)tok * DHEAD + d] = (bf16)(v * ATTN_SCALE);
        else if (which == 1)
          Kb[bh * HEADSZ + (size_t)tok * DHEAD + d] = (bf16)v;
        else
          Vt[bh * HEADSZ + (size_t)d * NTOK + tok] = (bf16)v;
      }
    }
  }
}

// ---------------------------------------------------------------------------
// Kernel 2: flash attention per (b,h, 128-query block).
//   256 threads = 8 waves; wave w owns query rows 16w..16w+15.
//   Per key block of 128: S = Q*K^T (WMMA) + bias, online softmax,
//   P -> LDS (C-layout -> A-layout reshape), O += P*V (WMMA).
//   K/V tiles async-staged to LDS when the toolchain supports it.
// ---------------------------------------------------------------------------
#define LDK 72    // K tile stride (bf16): 144B rows
#define LDV 136   // Vt tile stride (bf16): 272B rows
#define LDP 72    // P tile stride (bf16)

__global__ __launch_bounds__(256)
void k_attn(const bf16* __restrict__ Qb, const bf16* __restrict__ Kb,
            const bf16* __restrict__ Vt, const float* __restrict__ bias,
            bf16* __restrict__ AO) {
  __shared__ alignas(16) bf16 Ks[128 * LDK];      // K tile  [token][d]
  __shared__ alignas(16) bf16 Vs[64 * LDV];       // Vt tile [d][token]
  __shared__ alignas(16) bf16 Ps[8 * 16 * LDP];   // per-wave P half-tile

  const int t    = threadIdx.x;
  const int lane = t & 31, w = t >> 5;
  const int r16  = lane & 15, hf = lane >> 4;
  const int bh   = blockIdx.x >> 3;
  const int qb   = blockIdx.x & 7;
  const int b    = bh / NHEADS;
  const int h    = bh - b * NHEADS;
  const size_t headoff = (size_t)bh * HEADSZ;

  // Q A-fragments straight from global: lane r16 reads its own row (contig d).
  Frag qf[2];
#pragma unroll
  for (int kc = 0; kc < 2; ++kc) {
    const bf16* qrow = Qb + headoff + (size_t)(qb * 128 + 16 * w + r16) * DHEAD;
    qf[kc].h[0] = *(const v8bf*)(qrow + kc * 32 + 8 * hf);
    qf[kc].h[1] = *(const v8bf*)(qrow + kc * 32 + 16 + 8 * hf);
  }

  float mrun[8], lrun[8];
#pragma unroll
  for (int r = 0; r < 8; ++r) { mrun[r] = -1e30f; lrun[r] = 0.f; }
  v8f Of[4];
#pragma unroll
  for (int of = 0; of < 4; ++of) Of[of] = vzero8();

  const float* bptr = bias + (size_t)h * NTOK * NTOK;
  bf16* Pw = Ps + w * 16 * LDP;

  for (int j = 0; j < 8; ++j) {
    // Stage K tile (128x64) and Vt tile (64x128).
#pragma unroll
    for (int i = 0; i < 4; ++i) {
      int id = t + 256 * i;
      int r  = id >> 3, c8 = (id & 7) * 8;
      stage16B(Kb + headoff + (size_t)(j * 128 + r) * DHEAD + c8,
               &Ks[r * LDK + c8]);
    }
#pragma unroll
    for (int i = 0; i < 4; ++i) {
      int id = t + 256 * i;
      int d  = id >> 4, c8 = (id & 15) * 8;
      stage16B(Vt + headoff + (size_t)d * NTOK + j * 128 + c8,
               &Vs[d * LDV + c8]);
    }
    stage_wait();
    __syncthreads();

    // S = Q * K^T : 8 col-tiles x 2 k-chunks.
    v8f S[8];
#pragma unroll
    for (int nf = 0; nf < 8; ++nf) S[nf] = vzero8();
#pragma unroll
    for (int nf = 0; nf < 8; ++nf) {
#pragma unroll
      for (int kc = 0; kc < 2; ++kc) {
        Frag bf_;
        int col = nf * 16 + r16;  // key token within tile
        bf_.h[0] = *(const v8bf*)&Ks[col * LDK + kc * 32 + 8 * hf];
        bf_.h[1] = *(const v8bf*)&Ks[col * LDK + kc * 32 + 16 + 8 * hf];
        S[nf] = wmma_bf16(qf[kc].v, bf_.v, S[nf]);
      }
    }

    // + additive position bias (scale already folded into Q).
#pragma unroll
    for (int nf = 0; nf < 8; ++nf) {
      int gcol = j * 128 + nf * 16 + r16;
#pragma unroll
      for (int r = 0; r < 8; ++r) {
        int grow = qb * 128 + 16 * w + r + 8 * hf;
        S[nf][r] += bptr[(size_t)grow * NTOK + gcol];
      }
    }

    // Online softmax. Row (r + 8*hf) lives in vgpr r across the 16-lane half.
    float nm[8];
#pragma unroll
    for (int r = 0; r < 8; ++r) {
      nm[r] = mrun[r];
#pragma unroll
      for (int nf = 0; nf < 8; ++nf) nm[r] = fmaxf(nm[r], S[nf][r]);
#pragma unroll
      for (int m = 1; m < 16; m <<= 1)
        nm[r] = fmaxf(nm[r], __shfl_xor(nm[r], m, 32));
    }
    float alpha[8], rs[8];
#pragma unroll
    for (int r = 0; r < 8; ++r) {
      alpha[r] = __expf(mrun[r] - nm[r]);
      mrun[r]  = nm[r];
      rs[r]    = 0.f;
    }
#pragma unroll
    for (int nf = 0; nf < 8; ++nf)
#pragma unroll
      for (int r = 0; r < 8; ++r) {
        S[nf][r] = __expf(S[nf][r] - mrun[r]);
        rs[r] += S[nf][r];
      }
#pragma unroll
    for (int r = 0; r < 8; ++r) {
#pragma unroll
      for (int m = 1; m < 16; m <<= 1) rs[r] += __shfl_xor(rs[r], m, 32);
      lrun[r] = lrun[r] * alpha[r] + rs[r];
    }
#pragma unroll
    for (int of = 0; of < 4; ++of)
#pragma unroll
      for (int r = 0; r < 8; ++r) Of[of][r] *= alpha[r];

    // O += P * V, in two 64-column halves through the per-wave P buffer.
    // (LDS ops from one wave stay in order; no barrier needed for Pw.)
#pragma unroll
    for (int half = 0; half < 2; ++half) {
#pragma unroll
      for (int nf2 = 0; nf2 < 4; ++nf2) {
        int nf = half * 4 + nf2;
#pragma unroll
        for (int r = 0; r < 8; ++r)
          Pw[(r + 8 * hf) * LDP + nf2 * 16 + r16] = (bf16)S[nf][r];
      }
#pragma unroll
      for (int kc2 = 0; kc2 < 2; ++kc2) {
        int kc = half * 2 + kc2;  // token chunk within the 128-key block
        Frag pa;
        pa.h[0] = *(const v8bf*)&Pw[r16 * LDP + kc2 * 32 + 8 * hf];
        pa.h[1] = *(const v8bf*)&Pw[r16 * LDP + kc2 * 32 + 16 + 8 * hf];
#pragma unroll
        for (int of = 0; of < 4; ++of) {
          Frag vb;
          int dcol = of * 16 + r16;
          vb.h[0] = *(const v8bf*)&Vs[dcol * LDV + kc * 32 + 8 * hf];
          vb.h[1] = *(const v8bf*)&Vs[dcol * LDV + kc * 32 + 16 + 8 * hf];
          Of[of] = wmma_bf16(pa.v, vb.v, Of[of]);
        }
      }
    }
    __syncthreads();
  }

  // Normalize and write attention output as bf16 [b*1024+tok][h*64+d].
#pragma unroll
  for (int of = 0; of < 4; ++of) {
#pragma unroll
    for (int r = 0; r < 8; ++r) {
      int grow = qb * 128 + 16 * w + r + 8 * hf;
      float v = Of[of][r] / lrun[r];
      AO[((size_t)b * NTOK + grow) * DIMC + h * DHEAD + of * 16 + r16] = (bf16)v;
    }
  }
}

// ---------------------------------------------------------------------------
// Kernel 3: output projection. out[16384,768] = AO(bf16) @ Wout + b_out (fp32)
// ---------------------------------------------------------------------------
__global__ __launch_bounds__(256)
void k_out_gemm(const bf16* __restrict__ A, const bf16* __restrict__ Wt,
                const float* __restrict__ bvec, float* __restrict__ Cout) {
  __shared__ alignas(16) bf16 As[128 * LDT];
  __shared__ alignas(16) bf16 Bs[128 * LDT];

  const int t    = threadIdx.x;
  const int lane = t & 31, w = t >> 5;
  const int wm   = w & 3, wn = w >> 2;
  const int r16  = lane & 15, hf = lane >> 4;
  const int blockRow = blockIdx.x * 128;
  const int blockCol = blockIdx.y * 128;

  v8f acc[2][4];
#pragma unroll
  for (int mi = 0; mi < 2; ++mi)
#pragma unroll
    for (int ni = 0; ni < 4; ++ni) acc[mi][ni] = vzero8();

  for (int kk = 0; kk < DIMC / 32; ++kk) {
    const int k0 = kk * 32;
#pragma unroll
    for (int i = 0; i < 2; ++i) {
      int id = t + 256 * i;
      int r  = id >> 2, c8 = (id & 3) * 8;
      stage16B(A + (size_t)(blockRow + r) * DIMC + k0 + c8, &As[r * LDT + c8]);
    }
#pragma unroll
    for (int i = 0; i < 2; ++i) {
      int id = t + 256 * i;
      int n  = id >> 2, c8 = (id & 3) * 8;
      stage16B(Wt + (size_t)(blockCol + n) * DIMC + k0 + c8, &Bs[n * LDT + c8]);
    }
    stage_wait();
    __syncthreads();

    Frag a[2], b[4];
#pragma unroll
    for (int mi = 0; mi < 2; ++mi) {
      int row = wm * 32 + mi * 16 + r16;
      a[mi].h[0] = *(const v8bf*)&As[row * LDT + 8 * hf];
      a[mi].h[1] = *(const v8bf*)&As[row * LDT + 16 + 8 * hf];
    }
#pragma unroll
    for (int ni = 0; ni < 4; ++ni) {
      int col = wn * 64 + ni * 16 + r16;
      b[ni].h[0] = *(const v8bf*)&Bs[col * LDT + 8 * hf];
      b[ni].h[1] = *(const v8bf*)&Bs[col * LDT + 16 + 8 * hf];
    }
#pragma unroll
    for (int mi = 0; mi < 2; ++mi)
#pragma unroll
      for (int ni = 0; ni < 4; ++ni)
        acc[mi][ni] = wmma_bf16(a[mi].v, b[ni].v, acc[mi][ni]);
    __syncthreads();
  }

#pragma unroll
  for (int mi = 0; mi < 2; ++mi) {
#pragma unroll
    for (int ni = 0; ni < 4; ++ni) {
      int gc = blockCol + wn * 64 + ni * 16 + r16;
      float bias = bvec[gc];
#pragma unroll
      for (int r = 0; r < 8; ++r) {
        int gr = blockRow + wm * 32 + mi * 16 + r + 8 * hf;
        Cout[(size_t)gr * DIMC + gc] = acc[mi][ni][r] + bias;
      }
    }
  }
}

// ---------------------------------------------------------------------------
// Launch
// ---------------------------------------------------------------------------
extern "C" void kernel_launch(void* const* d_in, const int* in_sizes, int n_in,
                              void* d_out, int out_size, void* d_ws,
                              size_t ws_size, hipStream_t stream) {
  const float* x         = (const float*)d_in[0];  // [16,1024,768]
  const float* w_qkv     = (const float*)d_in[1];  // [768,2304]
  const float* attn_bias = (const float*)d_in[2];  // [12,1024,1024]
  const float* w_out     = (const float*)d_in[3];  // [768,768]
  const float* b_out     = (const float*)d_in[4];  // [768]
  float* out = (float*)d_out;

  char* ws = (char*)d_ws;
  size_t off = 0;
  auto carve = [&](size_t bytes) {
    void* p = ws + off;
    off += (bytes + 255) & ~(size_t)255;
    return p;
  };
  bf16* WqkvT = (bf16*)carve((size_t)QKVCOLS * DIMC * sizeof(bf16));
  bf16* WoutT = (bf16*)carve((size_t)DIMC * DIMC * sizeof(bf16));
  bf16* Qb    = (bf16*)carve((size_t)NBH * HEADSZ * sizeof(bf16));
  bf16* Kb    = (bf16*)carve((size_t)NBH * HEADSZ * sizeof(bf16));
  bf16* Vtb   = (bf16*)carve((size_t)NBH * HEADSZ * sizeof(bf16));
  bf16* AO    = (bf16*)carve((size_t)TOTROWS * DIMC * sizeof(bf16));

  // 1) weight prep: fp32 [K,N] -> bf16 [N,K]
  {
    int n1 = DIMC * QKVCOLS;
    k_transpose_bf16<<<(n1 + 255) / 256, 256, 0, stream>>>(w_qkv, WqkvT, DIMC, QKVCOLS);
    int n2 = DIMC * DIMC;
    k_transpose_bf16<<<(n2 + 255) / 256, 256, 0, stream>>>(w_out, WoutT, DIMC, DIMC);
  }
  // 2) QKV projection -> per-head Q (scaled), K, V^T
  {
    dim3 grid(TOTROWS / 128, QKVCOLS / 128);
    k_qkv_gemm<<<grid, 256, 0, stream>>>(x, WqkvT, Qb, Kb, Vtb);
  }
  // 3) flash attention
  {
    dim3 grid(NBH * (NTOK / 128));
    k_attn<<<grid, 256, 0, stream>>>(Qb, Kb, Vtb, attn_bias, AO);
  }
  // 4) output projection + bias
  {
    dim3 grid(TOTROWS / 128, DIMC / 128);
    k_out_gemm<<<grid, 256, 0, stream>>>(AO, WoutT, b_out, out);
  }
}
